// deform_conv2d_naive_20710332301391
// MI455X (gfx1250) — compile-verified, matrix operand order checked by reading, not run on hardware
//
#include <hip/hip_runtime.h>

// ---- CDNA5 WMMA vector types (wave32) ----
typedef __attribute__((ext_vector_type(16))) _Float16 v16h;
typedef __attribute__((ext_vector_type(8)))  _Float16 v8h;
typedef __attribute__((ext_vector_type(4)))  _Float16 v4h;
typedef __attribute__((ext_vector_type(2)))  _Float16 v2h;
typedef __attribute__((ext_vector_type(8)))  float    v8f;

#define C_IN   256
#define OC_ALL 256
#define H_IN   80
#define W_IN   80
#define NPIX   (H_IN * W_IN)      // 6400
#define KK     9
#define KTOT   (C_IN * KK)        // 2304
#define KQ4    (KTOT / 4)         // 576 float4s per weight row
#define KT     32                 // K chunk (16x16x32 f16 WMMA)
#define MT     256                // OC tile per block = ALL output channels
#define NT     32                 // pixel tile per block

__device__ __forceinline__ int clampi(int v, int lo, int hi) {
    return v < lo ? lo : (v > hi ? hi : v);
}

__global__ __launch_bounds__(256)
void deform_conv2d_wmma_kernel(const float* __restrict__ x,
                               const float* __restrict__ off,
                               const float* __restrict__ wgt,
                               float* __restrict__ out)
{
    // Double-buffered staging (ping-pong): 1 barrier per K-chunk
    __shared__ __align__(16) _Float16 Wbuf[2][MT][KT];   // 32 KB  [buf][oc][k]
    __shared__ __align__(16) _Float16 Sbuf[2][NT][KT];   // 4 KB   [buf][pixel][k]
    __shared__ float4 Tw[KK * NT];                       // 4.5 KB corner weights
    __shared__ int4   Ti[KK * NT];                       // 4.5 KB corner indices

    const int tid  = threadIdx.x;
    const int lane = tid & 31;
    const int wv   = tid >> 5;                 // 0..7 (8 waves)
    const int pt   = blockIdx.x;               // pixel tile (0..199)
    const int n    = blockIdx.y;               // image

    const int p_base = pt * NT;
    const float*  xn   = x   + (size_t)n * C_IN * NPIX;   // uniform -> SGPR base
    const float*  offn = off + (size_t)n * (2 * KK) * NPIX;
    float*        outn = out + (size_t)n * OC_ALL * NPIX;
    const float4* wgt4 = (const float4*)wgt;              // float4-granular indexing

    // ---- Phase 0: bilinear sampling table, 9 taps x 32 pixels ----
    for (int e = tid; e < KK * NT; e += 256) {
        const int t  = e >> 5;                 // tap 0..8 (ky*3+kx)
        const int p  = e & 31;
        const int pg = p_base + p;
        const int r  = pg / W_IN;
        const int cc = pg - r * W_IN;
        const int ky = t / 3;
        const int kx = t - 3 * ky;

        const float oy = offn[(uint)(2 * t)     * NPIX + (uint)pg];
        const float ox = offn[(uint)(2 * t + 1) * NPIX + (uint)pg];
        const float py = oy + (float)(ky + r - 1);   // stride 1, pad 1, dil 1
        const float px = ox + (float)(kx + cc - 1);

        const float fy0 = floorf(py);
        const float fx0 = floorf(px);
        const float ay  = py - fy0;
        const float ax  = px - fx0;
        const int y0 = (int)fy0, x0 = (int)fx0;
        const int y1 = y0 + 1,   x1 = x0 + 1;

        const float my0 = (y0 >= 0 && y0 < H_IN) ? 1.0f : 0.0f;
        const float my1 = (y1 >= 0 && y1 < H_IN) ? 1.0f : 0.0f;
        const float mx0 = (x0 >= 0 && x0 < W_IN) ? 1.0f : 0.0f;
        const float mx1 = (x1 >= 0 && x1 < W_IN) ? 1.0f : 0.0f;

        const int y0c = clampi(y0, 0, H_IN - 1);
        const int y1c = clampi(y1, 0, H_IN - 1);
        const int x0c = clampi(x0, 0, W_IN - 1);
        const int x1c = clampi(x1, 0, W_IN - 1);

        Tw[e] = make_float4((1.0f - ax) * (1.0f - ay) * mx0 * my0,
                            ax          * (1.0f - ay) * mx1 * my0,
                            (1.0f - ax) * ay          * mx0 * my1,
                            ax          * ay          * mx1 * my1);
        Ti[e] = make_int4(y0c * W_IN + x0c, y0c * W_IN + x1c,
                          y1c * W_IN + x0c, y1c * W_IN + x1c);
    }

    // Wave's output strip: 64 rows (4 x 16) x 16 pixels
    const int mw = (wv >> 1) * 64;             // 0,64,128,192
    const int n0 = (wv & 1) * 16;              // 0,16
    const int hl = lane & 15;
    const int hs = lane >> 4;                  // half-select K{0..7,16..23} / K{8..15,24..31}

    v8f acc0 = {}, acc1 = {}, acc2 = {}, acc3 = {};
    __syncthreads();

    for (int kb = 0; kb < KTOT; kb += KT) {
        const int buf = (kb >> 5) & 1;

        // ---- stage weight chunk: 256 x 32 f16 (b128 scale_offset loads) ----
        #pragma unroll
        for (int i = 0; i < 8; ++i) {
            const int q  = tid + 256 * i;      // quad id (0..2047)
            const int m  = q >> 3;             // oc row
            const int kq = (q & 7) << 2;       // k quad within chunk
            const float4 w4 = wgt4[(uint)(m * KQ4 + (kb >> 2) + (q & 7))];
            v4h h;
            h[0] = (_Float16)w4.x; h[1] = (_Float16)w4.y;
            h[2] = (_Float16)w4.z; h[3] = (_Float16)w4.w;
            *(v4h*)&Wbuf[buf][m][kq] = h;
        }
        // next chunk's first row: always in-bounds (row stride 2304 >= kb+32)
        __builtin_prefetch(wgt + (uint)(kb + KT), 0, 1);

        // ---- stage sampled chunk: k-pairs -> packed cvt + ds_store_b32 ----
        #pragma unroll
        for (int i = 0; i < 2; ++i) {
            const int e  = tid + 256 * i;      // pair id (0..511)
            const int p  = e & 31;             // pixel within tile
            const int k2 = (e >> 5) << 1;      // even k within chunk
            float vv[2];
            #pragma unroll
            for (int u = 0; u < 2; ++u) {
                const int kg = kb + k2 + u;
                const int c  = kg / 9;
                const int t  = kg - 9 * c;
                const float4 w4 = Tw[t * NT + p];
                const int4   i4 = Ti[t * NT + p];
                const uint cb = (uint)c * NPIX;
                vv[u] = w4.x * xn[cb + (uint)i4.x] + w4.y * xn[cb + (uint)i4.y]
                      + w4.z * xn[cb + (uint)i4.z] + w4.w * xn[cb + (uint)i4.w];
            }
            v2h h2;
            h2[0] = (_Float16)vv[0];
            h2[1] = (_Float16)vv[1];
            *(v2h*)&Sbuf[buf][p][k2] = h2;
        }
        __syncthreads();   // staging(buf) done -> compute(buf); prev buf now free

        // ---- B fragment once, then 4 A fragments + 4 WMMAs ----
        const v8h b0 = *(const v8h*)&Sbuf[buf][n0 + hl][hs * 8];
        const v8h b1 = *(const v8h*)&Sbuf[buf][n0 + hl][16 + hs * 8];
        v16h b;
        #pragma unroll
        for (int i = 0; i < 8; ++i) { b[i] = b0[i]; b[8 + i] = b1[i]; }

        #pragma unroll
        for (int j = 0; j < 4; ++j) {
            const int row = mw + j * 16 + hl;
            const v8h a0 = *(const v8h*)&Wbuf[buf][row][hs * 8];
            const v8h a1 = *(const v8h*)&Wbuf[buf][row][16 + hs * 8];
            v16h a;
            #pragma unroll
            for (int i = 0; i < 8; ++i) { a[i] = a0[i]; a[8 + i] = a1[i]; }
            v8f* accp = (j == 0) ? &acc0 : (j == 1) ? &acc1 : (j == 2) ? &acc2 : &acc3;
            *accp = __builtin_amdgcn_wmma_f32_16x16x32_f16(
                false, a, false, b, (short)0, *accp, false, false);
        }
        // no trailing barrier: ds reads completed before WMMA (s_wait_dscnt),
        // and next iteration writes the other buffer
    }

    // ---- store: D tile layout VGPR r -> M = r + 8*hs, N = hl ----
    const uint pcol = (uint)(p_base + n0 + hl);
    #pragma unroll
    for (int j = 0; j < 4; ++j) {
        const v8f acc = (j == 0) ? acc0 : (j == 1) ? acc1 : (j == 2) ? acc2 : acc3;
        #pragma unroll
        for (int rr = 0; rr < 8; ++rr) {
            const int m = mw + j * 16 + hs * 8 + rr;
            outn[(uint)m * NPIX + pcol] = acc[rr];
        }
    }
}

extern "C" void kernel_launch(void* const* d_in, const int* in_sizes, int n_in,
                              void* d_out, int out_size, void* d_ws, size_t ws_size,
                              hipStream_t stream) {
    (void)in_sizes; (void)n_in; (void)out_size; (void)d_ws; (void)ws_size;
    const float* x   = (const float*)d_in[0];   // (4,256,80,80)
    const float* off = (const float*)d_in[1];   // (4,18,80,80)
    const float* wgt = (const float*)d_in[2];   // (256,256,3,3) -> (256,2304)
    float* out = (float*)d_out;                 // (4,256,80,80)

    dim3 grid(NPIX / NT, 4);                    // (200 pixel tiles, 4 images)
    dim3 block(256);                            // 8 waves (wave32)
    deform_conv2d_wmma_kernel<<<grid, block, 0, stream>>>(x, off, wgt, out);
}
